// Attention_24885040513035
// MI455X (gfx1250) — matrix operand order repaired and checked by reading, not compiled, not run on hardware
//
#include <hip/hip_runtime.h>

// ---------------------------------------------------------------------------
// corr_attention for MI455X (gfx1250, wave32, WMMA)
// B=4 H=8 S=1024 D=64  ->  BH=32, n = 2^25 global scores
// ---------------------------------------------------------------------------

typedef __bf16 bf16;
typedef __attribute__((ext_vector_type(16))) __bf16 v16bf;
typedef __attribute__((ext_vector_type(8)))  float  v8f;

#define S_LEN   1024
#define D_DIM   64
#define BH_CNT  32
#define N_ELEM  33554432u          // 2^25 scores
#define NBINS   (1u << 21)         // key>>9 for |s|<2.0 fits in 2^21 bins
#define KEY_SHIFT 9

// ---------------------------------------------------------------------------
// 1) Per-row L2 norm folded into value, split fp32 -> bf16 hi + lo.
//    mode 1: normalize rows (Q, K), layout [bh][s][d]
//    mode 2: no normalize, TRANSPOSED output [bh][d][s]  (V for gemm2 B-side)
//    one wave per row of 64 floats (2 per lane)
// ---------------------------------------------------------------------------
__global__ __launch_bounds__(256)
void prep_split_kernel(const float* __restrict__ src,
                       bf16* __restrict__ hi, bf16* __restrict__ lo,
                       int mode) {
  int w    = threadIdx.x >> 5;
  int lane = threadIdx.x & 31;
  int row  = blockIdx.x * 8 + w;                 // 0 .. 32767  (= bh*1024 + s)
  size_t o = (size_t)row * D_DIM + lane * 2;
  float x0 = src[o], x1 = src[o + 1];
  float inv = 1.0f;
  if (mode == 1) {
    float ss = x0 * x0 + x1 * x1;
    #pragma unroll
    for (int m = 16; m >= 1; m >>= 1) ss += __shfl_xor(ss, m, 32);
    inv = 1.0f / (sqrtf(ss) + 1e-5f);
  }
  float y0 = x0 * inv, y1 = x1 * inv;
  bf16 h0 = (bf16)y0, h1 = (bf16)y1;
  bf16 l0 = (bf16)(y0 - (float)h0), l1 = (bf16)(y1 - (float)h1);
  if (mode == 2) {
    int bh = row >> 10, s = row & 1023, d = lane * 2;
    size_t t0 = ((size_t)bh * D_DIM + d) * S_LEN + s;
    hi[t0] = h0;            lo[t0] = l0;
    hi[t0 + S_LEN] = h1;    lo[t0 + S_LEN] = l1;
  } else {
    hi[o]     = h0;  lo[o]     = l0;
    hi[o + 1] = h1;  lo[o + 1] = l1;
  }
}

// ---------------------------------------------------------------------------
// 2) score = Qn * Kn^T  (cosine similarity), 3x bf16-split WMMA per K-step.
//    One wave owns a 16x16 tile; K-loop over D=64 in two 16x16x32 steps.
// ---------------------------------------------------------------------------
__global__ __launch_bounds__(256)
void gemm1_kernel(const bf16* __restrict__ Qhi, const bf16* __restrict__ Qlo,
                  const bf16* __restrict__ Khi, const bf16* __restrict__ Klo,
                  float* __restrict__ score) {
  int wave = threadIdx.x >> 5;
  int lane = threadIdx.x & 31;
  int tile = blockIdx.x * 8 + wave;              // 0 .. 131071
  int bh   = tile >> 12;                         // / (64*64)
  int rem  = tile & 4095;
  int i0   = (rem >> 6) * 16;
  int j0   = (rem & 63) * 16;
  int m    = lane & 15;
  int half = lane >> 4;                          // 0: lanes 0-15, 1: 16-31
  int n    = lane & 15;

  const bf16* arh = Qhi + ((size_t)bh * S_LEN + i0 + m) * D_DIM;
  const bf16* arl = Qlo + ((size_t)bh * S_LEN + i0 + m) * D_DIM;
  const bf16* bch = Khi + ((size_t)bh * S_LEN + j0 + n) * D_DIM;
  const bf16* bcl = Klo + ((size_t)bh * S_LEN + j0 + n) * D_DIM;

  v8f c = {};
  #pragma unroll
  for (int k0 = 0; k0 < D_DIM; k0 += 32) {
    // A (16x32 bf16): lane half selects K-subgroup; elems 0-7 -> K=kb..kb+7,
    // elems 8-15 -> K=kb+16..kb+23  (ISA 16-bit A layout)
    int kbA = k0 + half * 8;
    v16bf ah, al;
    #pragma unroll
    for (int e = 0; e < 8; ++e) {
      ah[e]     = arh[kbA + e];       al[e]     = arl[kbA + e];
      ah[8 + e] = arh[kbA + 16 + e];  al[8 + e] = arl[kbA + 16 + e];
    }
    // B (32x16 bf16): column n = lane&15, same K order as A: elems 0-7 -> K=kb..kb+7,
    // elems 8-15 -> K=kb+16..kb+23
    int kbB = k0 + half * 8;
    v16bf bh_, bl_;
    #pragma unroll
    for (int e = 0; e < 8; ++e) {
      bh_[e]     = bch[kbB + e];       bl_[e]     = bcl[kbB + e];
      bh_[8 + e] = bch[kbB + 16 + e];  bl_[8 + e] = bcl[kbB + 16 + e];
    }

    c = __builtin_amdgcn_wmma_f32_16x16x32_bf16(false, ah, false, bh_, (short)0, c, false, false);
    c = __builtin_amdgcn_wmma_f32_16x16x32_bf16(false, ah, false, bl_, (short)0, c, false, false);
    c = __builtin_amdgcn_wmma_f32_16x16x32_bf16(false, al, false, bh_, (short)0, c, false, false);
  }
  // C/D layout: VGPR v -> row i0 + v + 8*half, col j0 + (lane&15)
  #pragma unroll
  for (int v = 0; v < 8; ++v) {
    int row = i0 + v + 8 * half;
    score[((size_t)bh * S_LEN + row) * S_LEN + j0 + n] = c[v];
  }
}

// ---------------------------------------------------------------------------
// 3) Global counting rank: histogram of key = bits(|s|) >> 9
// ---------------------------------------------------------------------------
__global__ __launch_bounds__(256)
void hist_zero_kernel(unsigned* __restrict__ hist) {
  size_t i = ((size_t)blockIdx.x * 256 + threadIdx.x) * 4;
  uint4 z = {0u, 0u, 0u, 0u};
  *(uint4*)(hist + i) = z;
}

__global__ __launch_bounds__(256)
void hist_build_kernel(const float* __restrict__ score, unsigned* __restrict__ hist) {
  size_t i = ((size_t)blockIdx.x * 256 + threadIdx.x) * 8;
  float4 a = *(const float4*)(score + i);
  float4 b = *(const float4*)(score + i + 4);
  float v[8] = {a.x, a.y, a.z, a.w, b.x, b.y, b.z, b.w};
  #pragma unroll
  for (int e = 0; e < 8; ++e) {
    unsigned bin = __float_as_uint(fabsf(v[e])) >> KEY_SHIFT;
    if (bin >= NBINS) bin = NBINS - 1;
    atomicAdd(&hist[bin], 1u);
  }
}

// ---- exclusive scan over 2^21 bins: partials -> top scan -> apply ---------
__global__ __launch_bounds__(256)
void scan_partials_kernel(const unsigned* __restrict__ hist, unsigned* __restrict__ partials) {
  __shared__ unsigned red[256];
  int t = threadIdx.x;
  size_t base = (size_t)blockIdx.x * 1024 + t * 4;
  red[t] = hist[base] + hist[base + 1] + hist[base + 2] + hist[base + 3];
  __syncthreads();
  for (int off = 128; off > 0; off >>= 1) {
    if (t < off) red[t] += red[t + off];
    __syncthreads();
  }
  if (t == 0) partials[blockIdx.x] = red[0];
}

__global__ __launch_bounds__(256)
void scan_top_kernel(unsigned* __restrict__ partials) {   // 2048 entries, in place
  __shared__ unsigned tmp[256];
  int t = threadIdx.x;
  unsigned loc[8];
  unsigned sum = 0;
  #pragma unroll
  for (int j = 0; j < 8; ++j) { loc[j] = partials[t * 8 + j]; sum += loc[j]; }
  tmp[t] = sum; __syncthreads();
  for (int off = 1; off < 256; off <<= 1) {
    unsigned x = (t >= off) ? tmp[t - off] : 0u;
    __syncthreads();
    tmp[t] += x;
    __syncthreads();
  }
  unsigned run = tmp[t] - sum;                            // exclusive
  #pragma unroll
  for (int j = 0; j < 8; ++j) { unsigned cnt = loc[j]; partials[t * 8 + j] = run; run += cnt; }
}

__global__ __launch_bounds__(256)
void scan_apply_kernel(unsigned* __restrict__ hist, const unsigned* __restrict__ partials) {
  __shared__ unsigned tmp[256];
  int t = threadIdx.x;
  size_t base = (size_t)blockIdx.x * 1024 + t * 4;
  unsigned c0 = hist[base], c1 = hist[base + 1], c2 = hist[base + 2], c3 = hist[base + 3];
  unsigned s = c0 + c1 + c2 + c3;
  tmp[t] = s; __syncthreads();
  for (int off = 1; off < 256; off <<= 1) {
    unsigned x = (t >= off) ? tmp[t - off] : 0u;
    __syncthreads();
    tmp[t] += x;
    __syncthreads();
  }
  unsigned excl = tmp[t] - s + partials[blockIdx.x];
  hist[base]     = excl;
  hist[base + 1] = excl + c0;
  hist[base + 2] = excl + c0 + c1;
  hist[base + 3] = excl + c0 + c1 + c2;
}

// ---------------------------------------------------------------------------
// 4) rank -> val = -log(rank/(n-1) + 1/n) * sign(s); row sum of |val|.
//    Deterministic: rank = base[bin] + cnt * low9 / 512 (interpolated).
//    One block per attention row (1024 elems).
// ---------------------------------------------------------------------------
__global__ __launch_bounds__(256)
void rank_rowsum_kernel(float* __restrict__ score, const unsigned* __restrict__ base,
                        float* __restrict__ rowsum) {
  __shared__ float red[256];
  int t = threadIdx.x;
  size_t row = blockIdx.x;
  float4* p = (float4*)(score + row * S_LEN) + t;
  float4 v = *p;
  const float inv_nm1 = 1.0f / (float)(N_ELEM - 1u);
  const float inv_n   = 1.0f / (float)N_ELEM;
  float in[4] = {v.x, v.y, v.z, v.w};
  float out[4];
  float acc = 0.0f;
  #pragma unroll
  for (int e = 0; e < 4; ++e) {
    float s = in[e];
    unsigned key = __float_as_uint(fabsf(s));
    unsigned bin = key >> KEY_SHIFT;
    if (bin >= NBINS) bin = NBINS - 1;
    unsigned b0  = base[bin];
    unsigned b1  = (bin + 1 < NBINS) ? base[bin + 1] : N_ELEM;
    unsigned cnt = b1 - b0;
    unsigned low = key & ((1u << KEY_SHIFT) - 1u);
    unsigned long long r = (unsigned long long)b0 +
                           (((unsigned long long)cnt * low) >> KEY_SHIFT);
    float ranked = (float)r * inv_nm1 + inv_n;
    float sgn = (s > 0.0f) ? 1.0f : ((s < 0.0f) ? -1.0f : 0.0f);
    float val = -logf(ranked) * sgn;
    out[e] = val;
    acc += fabsf(val);
  }
  float4 o = {out[0], out[1], out[2], out[3]};
  *p = o;
  red[t] = acc; __syncthreads();
  for (int off = 128; off > 0; off >>= 1) {
    if (t < off) red[t] += red[t + off];
    __syncthreads();
  }
  if (t == 0) rowsum[row] = red[0];
}

// ---------------------------------------------------------------------------
// 5) out = (val / rowsum) @ V, 3x bf16-split WMMA, row scale at store.
//    B side reads the TRANSPOSED V copy [bh][d][j]: per-lane fragments are
//    contiguous 32-byte runs -> global_load_b128 instead of strided d16.
// ---------------------------------------------------------------------------
__global__ __launch_bounds__(256)
void gemm2_kernel(const float* __restrict__ val,
                  const bf16* __restrict__ Vthi, const bf16* __restrict__ Vtlo,
                  const float* __restrict__ rowsum,
                  float* __restrict__ out) {
  int wave = threadIdx.x >> 5;
  int lane = threadIdx.x & 31;
  int tile = blockIdx.x * 8 + wave;              // 0 .. 8191
  int bh   = tile >> 8;                          // / (64*4)
  int rem  = tile & 255;
  int i0   = (rem >> 2) * 16;
  int d0   = (rem & 3) * 16;
  int m    = lane & 15;
  int half = lane >> 4;
  int n    = lane & 15;

  const float* arow = val + ((size_t)bh * S_LEN + i0 + m) * S_LEN;
  const bf16* bph = Vthi + ((size_t)bh * D_DIM + d0 + n) * S_LEN + half * 16;
  const bf16* bpl = Vtlo + ((size_t)bh * D_DIM + d0 + n) * S_LEN + half * 16;

  v8f c = {};
  for (int j0 = 0; j0 < S_LEN; j0 += 32) {
    int kbA = j0 + half * 16;
    __builtin_prefetch(arow + kbA + 32, 0, 0);   // global_prefetch_b8
    float4 fa = *(const float4*)(arow + kbA);
    float4 fb = *(const float4*)(arow + kbA + 4);
    float4 fc = *(const float4*)(arow + kbA + 8);
    float4 fd = *(const float4*)(arow + kbA + 12);
    float af[16] = {fa.x, fa.y, fa.z, fa.w, fb.x, fb.y, fb.z, fb.w,
                    fc.x, fc.y, fc.z, fc.w, fd.x, fd.y, fd.z, fd.w};
    v16bf ah, al;
    #pragma unroll
    for (int e = 0; e < 16; ++e) {
      bf16 h = (bf16)af[e];
      ah[e] = h;
      al[e] = (bf16)(af[e] - (float)h);
    }
    // B fragment: column d0+n, K = j0 + half*16 + e  -> contiguous in Vt
    v16bf bh_, bl_;
    #pragma unroll
    for (int e = 0; e < 16; ++e) { bh_[e] = bph[j0 + e]; bl_[e] = bpl[j0 + e]; }

    c = __builtin_amdgcn_wmma_f32_16x16x32_bf16(false, ah, false, bh_, (short)0, c, false, false);
    c = __builtin_amdgcn_wmma_f32_16x16x32_bf16(false, ah, false, bl_, (short)0, c, false, false);
    c = __builtin_amdgcn_wmma_f32_16x16x32_bf16(false, al, false, bh_, (short)0, c, false, false);
  }
  #pragma unroll
  for (int v = 0; v < 8; ++v) {
    int row = i0 + v + 8 * half;
    float rs = rowsum[(size_t)bh * S_LEN + row];
    out[((size_t)bh * S_LEN + row) * D_DIM + d0 + n] = c[v] / rs;
  }
}

// ---------------------------------------------------------------------------
extern "C" void kernel_launch(void* const* d_in, const int* in_sizes, int n_in,
                              void* d_out, int out_size, void* d_ws, size_t ws_size,
                              hipStream_t stream) {
  (void)in_sizes; (void)n_in; (void)out_size; (void)ws_size;
  const float* Q = (const float*)d_in[0];
  const float* K = (const float*)d_in[1];
  const float* V = (const float*)d_in[2];
  float* out = (float*)d_out;

  // workspace layout (all 256B aligned)
  char* ws = (char*)d_ws;
  size_t off = 0;
  float*    score    = (float*)(ws + off);    off += (size_t)N_ELEM * 4;      // 128 MB
  unsigned* hist     = (unsigned*)(ws + off); off += (size_t)NBINS * 4;       // 8 MB
  unsigned* partials = (unsigned*)(ws + off); off += 2048 * 4;
  off = (off + 255) & ~(size_t)255;
  float*    rowsum   = (float*)(ws + off);    off += (size_t)BH_CNT * S_LEN * 4;
  const size_t nQ = (size_t)BH_CNT * S_LEN * D_DIM;                            // 2M elems
  bf16* Qhi  = (bf16*)(ws + off); off += nQ * 2;
  bf16* Qlo  = (bf16*)(ws + off); off += nQ * 2;
  bf16* Khi  = (bf16*)(ws + off); off += nQ * 2;
  bf16* Klo  = (bf16*)(ws + off); off += nQ * 2;
  bf16* Vthi = (bf16*)(ws + off); off += nQ * 2;   // transposed [bh][d][j]
  bf16* Vtlo = (bf16*)(ws + off); off += nQ * 2;

  // 1) normalize + bf16 hi/lo split (V: transposed copy for gemm2 B-side)
  prep_split_kernel<<<4096, 256, 0, stream>>>(Q, Qhi, Qlo, 1);
  prep_split_kernel<<<4096, 256, 0, stream>>>(K, Khi, Klo, 1);
  prep_split_kernel<<<4096, 256, 0, stream>>>(V, Vthi, Vtlo, 2);

  // 2) cosine-similarity scores via WMMA
  gemm1_kernel<<<16384, 256, 0, stream>>>(Qhi, Qlo, Khi, Klo, score);

  // 3) global counting rank (L2-resident passes)
  hist_zero_kernel<<<2048, 256, 0, stream>>>(hist);
  hist_build_kernel<<<16384, 256, 0, stream>>>(score, hist);
  scan_partials_kernel<<<2048, 256, 0, stream>>>(hist, partials);
  scan_top_kernel<<<1, 256, 0, stream>>>(partials);
  scan_apply_kernel<<<2048, 256, 0, stream>>>(hist, partials);

  // 4) -log(cdf)*sign + row |.| sums
  rank_rowsum_kernel<<<32768, 256, 0, stream>>>(score, hist, rowsum);

  // 5) normalized score @ V via WMMA
  gemm2_kernel<<<1024, 256, 0, stream>>>(score, Vthi, Vtlo, rowsum, out);
}
